// CTG_block_80857054314955
// MI455X (gfx1250) — compile-verified, matrix operand order checked by reading, not run on hardware
//
#include <hip/hip_runtime.h>
#include <hip/hip_bf16.h>

// MI455X / gfx1250 implementation: wave32, WMMA f16 (16x16x32) with f32 accum.
// Pipeline: [k_cvt weights->f16] -> [k_warp_ln1] -> [k_attn per window] -> [k_mlp per 64-token tile]
// Round 3: fix async-to-LDS builtin signature (takes v4i addrspace pointers, not void*).

#define DEV __device__ __forceinline__

typedef __attribute__((ext_vector_type(16))) _Float16 v16h;
typedef __attribute__((ext_vector_type(8)))  float    v8f;
typedef __attribute__((ext_vector_type(4)))  int      v4i;

#if __has_builtin(__builtin_amdgcn_global_load_async_to_lds_b128) && \
    __has_builtin(__builtin_amdgcn_s_wait_asynccnt)
#define HAVE_ASYNC_LDS 1
#endif

typedef __attribute__((address_space(1))) v4i* gv4i_p;
typedef __attribute__((address_space(3))) v4i* lv4i_p;

DEV void async_b128(const void* g, void* l) {
#ifdef HAVE_ASYNC_LDS
    __builtin_amdgcn_global_load_async_to_lds_b128(
        (gv4i_p)(g), (lv4i_p)(l), 0, 0);
#endif
}

DEV void async_wait0() {
#ifdef HAVE_ASYNC_LDS
    __builtin_amdgcn_s_wait_asynccnt(0);
#endif
}

DEV v8f wmma_f16(v16h a, v16h b, v8f c) {
    return __builtin_amdgcn_wmma_f32_16x16x32_f16(false, a, false, b, (short)0, c, false, false);
}

// A fragment: 16x32 f16, source row-major [16][ld], element (m,k) at p[m*ld+k].
// Wave32 layout: lane L -> M = L&15; halves i: K = kb + (i>>3)*16 + (i&7), kb = (L&16)?8:0.
DEV v16h load_a(const _Float16* __restrict__ p, int ld, int lane) {
    const int m  = lane & 15;
    const int kb = (lane & 16) ? 8 : 0;
    v16h a;
#pragma unroll
    for (int i = 0; i < 16; ++i) {
        const int k = kb + ((i >> 3) << 4) + (i & 7);
        a[i] = p[m * ld + k];
    }
    return a;
}

// B fragment: 32x16 f16 from row-major K-major source, element (k,n) at p[k*ld+n].
DEV v16h load_b(const _Float16* __restrict__ p, int ld, int lane) {
    const int n  = lane & 15;
    const int kb = (lane & 16) ? 8 : 0;
    v16h b;
#pragma unroll
    for (int i = 0; i < 16; ++i) {
        const int k = kb + ((i >> 3) << 4) + (i & 7);
        b[i] = p[k * ld + n];
    }
    return b;
}

// B fragment from a weight matrix W[N][K] row-major (i.e. B = W^T): element (k,n) at p[n*ld+k].
DEV v16h load_bt(const _Float16* __restrict__ p, int ld, int lane) {
    const int n  = lane & 15;
    const int kb = (lane & 16) ? 8 : 0;
    v16h b;
#pragma unroll
    for (int i = 0; i < 16; ++i) {
        const int k = kb + ((i >> 3) << 4) + (i & 7);
        b[i] = p[n * ld + k];
    }
    return b;
}

// ---------------------------------------------------------------------------
// Kernel 0: convert weight matrices f32 -> f16 once.
// layout in halves: qkv_w [0,12288) proj_w [12288,16384) fc1_w [16384,32768)
//                   fc2_w [32768,49152) conv_w [49152,51200)
__global__ void k_cvt(const float* __restrict__ qkv_w, const float* __restrict__ proj_w,
                      const float* __restrict__ fc1_w, const float* __restrict__ fc2_w,
                      const float* __restrict__ conv_w, _Float16* __restrict__ d) {
    const int i = blockIdx.x * 256 + threadIdx.x;
    if (i < 12288)      d[i] = (_Float16)qkv_w[i];
    else if (i < 16384) d[i] = (_Float16)proj_w[i - 12288];
    else if (i < 32768) d[i] = (_Float16)fc1_w[i - 16384];
    else if (i < 49152) d[i] = (_Float16)fc2_w[i - 32768];
    else if (i < 51200) d[i] = (_Float16)conv_w[i - 49152];
}

// ---------------------------------------------------------------------------
// Kernel 1: bilinear warp + concat + LN1.
// One block = one token (B*h*w = 460800 blocks), 64 threads = 64 channels.
// Writes: x  [token][64] f32 (shortcut),  xw [win][64 tok][64 ch] f16 (LN1 output, window layout)
__global__ void k_warp_ln1(const float* __restrict__ gfea, const float* __restrict__ cfea,
                           const float* __restrict__ disp, const float* __restrict__ g1,
                           const float* __restrict__ b1, float* __restrict__ x,
                           _Float16* __restrict__ xw) {
    const int token = blockIdx.x;
    const int ch    = threadIdx.x;
    const int Bi    = token / 9216;
    const int pos   = token - Bi * 9216;
    const int bb    = Bi / 25;
    const int ang   = Bi - bb * 25;
    const int y     = pos / 96;
    const int xc    = pos - y * 96;

    float val;
    if (ch < 32) {
        val = gfea[(size_t)(Bi * 32 + ch) * 9216 + pos];
    } else {
        const int   cc   = ch - 32;
        const float d    = disp[bb * 9216 + pos];
        const float offy = (float)(ang / 5 - 2);
        const float offx = (float)(ang % 5 - 2);
        const float sy   = (float)y + d * offy;
        const float sx   = (float)xc + d * offx;
        const float x0f  = floorf(sx), y0f = floorf(sy);
        const float wx   = sx - x0f,   wy  = sy - y0f;
        int x0 = (int)x0f; x0 = x0 < 0 ? 0 : (x0 > 95 ? 95 : x0);
        int x1 = x0 + 1;   x1 = x1 > 95 ? 95 : x1;
        int y0 = (int)y0f; y0 = y0 < 0 ? 0 : (y0 > 95 ? 95 : y0);
        int y1 = y0 + 1;   y1 = y1 > 95 ? 95 : y1;
        const float* f = cfea + (size_t)(bb * 32 + cc) * 9216;
        val = (1.f - wy) * (1.f - wx) * f[y0 * 96 + x0]
            + (1.f - wy) * wx         * f[y0 * 96 + x1]
            + wy         * (1.f - wx) * f[y1 * 96 + x0]
            + wy         * wx         * f[y1 * 96 + x1];
    }

    __shared__ float red[64];
    __shared__ float mv[2];
    red[ch] = val;
    __syncthreads();
    if (ch == 0) { float s = 0.f; for (int i = 0; i < 64; ++i) s += red[i]; mv[0] = s * (1.f / 64.f); }
    __syncthreads();
    const float m = mv[0];
    red[ch] = (val - m) * (val - m);
    __syncthreads();
    if (ch == 0) { float s = 0.f; for (int i = 0; i < 64; ++i) s += red[i]; mv[1] = s * (1.f / 64.f); }
    __syncthreads();
    const float rs = rsqrtf(mv[1] + 1e-5f);
    const float xn = (val - m) * rs * g1[ch] + b1[ch];

    x[(size_t)token * 64 + ch] = val;

    const int win = Bi * 144 + (y >> 3) * 12 + (xc >> 3);
    const int n   = ((y & 7) << 3) + (xc & 7);
    xw[((size_t)win * 64 + n) * 64 + ch] = (_Float16)xn;
}

// ---------------------------------------------------------------------------
// Kernel 2: window attention. One block per window, 128 threads (4 waves).
// 56KB LDS, phase-overlaid regions:
//   P1: sX [0,8K) + sW [8K,32K)  -> QKV gemm -> sQKV [32K,56K) (q pre-scaled)
//   P2: scores (QK^T + rpb)      -> sAttn f32 [0,32K)
//   P3: softmax                  -> sA16 f16 [32K,48K) (overlays dead q,k)
//   P4: async-load sPW [8K,16K) overlapped with AV gemm -> sO f16 [0,8K)
//   P5: proj gemm + bias + residual -> x (in place, f32)
__global__ __launch_bounds__(128)
void k_attn(const _Float16* __restrict__ xw, const _Float16* __restrict__ w16,
            const float* __restrict__ qkv_b, const float* __restrict__ proj_b,
            const float* __restrict__ rpb, float* __restrict__ x) {
    __shared__ __align__(16) char smem[57344];
    _Float16* sX   = (_Float16*)smem;                 // 64x64 f16 tokens
    _Float16* sW   = (_Float16*)(smem + 8192);        // 192x64 f16 qkv weights
    _Float16* sQKV = (_Float16*)(smem + 32768);       // q,k,v each 64x64 f16
    float*    sAttn= (float*)smem;                    // 2 heads x 64x64 f32
    _Float16* sA16 = (_Float16*)(smem + 32768);       // 2 heads x 64x64 f16 (over q,k)
    _Float16* sV   = (_Float16*)(smem + 49152);       // v region (persists)
    _Float16* sO   = (_Float16*)smem;                 // 64x64 f16 attn out
    _Float16* sPW  = (_Float16*)(smem + 8192);        // 64x64 f16 proj weights

    const _Float16* qkvw16  = w16;
    const _Float16* projw16 = w16 + 12288;

    const int tid  = threadIdx.x;
    const int wave = tid >> 5, lane = tid & 31;
    const int win  = blockIdx.x;

    // ---- P1: stage tokens + qkv weights (async direct-to-LDS when available)
#ifdef HAVE_ASYNC_LDS
    {
        const char* gx = (const char*)(xw + (size_t)win * 4096);
        for (int i = tid * 16; i < 8192; i += 128 * 16)
            async_b128(gx + i, (char*)sX + i);
        const char* gw = (const char*)qkvw16;
        for (int i = tid * 16; i < 24576; i += 128 * 16)
            async_b128(gw + i, (char*)sW + i);
        async_wait0();
    }
#else
    for (int i = tid; i < 4096;  i += 128) sX[i] = xw[(size_t)win * 4096 + i];
    for (int i = tid; i < 12288; i += 128) sW[i] = qkvw16[i];
#endif
    __syncthreads();

    // ---- QKV gemm: X(64x64) @ W^T(64x192)
    for (int t = wave; t < 48; t += 4) {
        const int mt = t & 3, nt = t >> 2;
        v8f acc = {};
#pragma unroll
        for (int kc = 0; kc < 64; kc += 32) {
            v16h a  = load_a (sX + mt * 16 * 64 + kc, 64, lane);
            v16h bf = load_bt(sW + (nt * 16) * 64 + kc, 64, lane);
            acc = wmma_f16(a, bf, acc);
        }
        const int   o    = nt * 16 + (lane & 15);
        const float bias = qkv_b[o];
        const float scl  = (o < 64) ? 0.17677669529663687f : 1.0f; // 1/sqrt(32) on q
        const int   mb   = (lane & 16) ? 8 : 0;
#pragma unroll
        for (int r = 0; r < 8; ++r) {
            const int row = mt * 16 + mb + r;
            sQKV[(o >> 6) * 4096 + row * 64 + (o & 63)] = (_Float16)((acc[r] + bias) * scl);
        }
    }
    __syncthreads();

    // ---- scores: per head q(64x32) @ k^T(32x64) + rel-pos bias
    for (int t = wave; t < 32; t += 4) {
        const int head = t >> 4, mt = (t >> 2) & 3, nt = t & 3;
        v16h a  = load_a (sQKV + mt * 16 * 64 + head * 32, 64, lane);
        v16h bf = load_bt(sQKV + 4096 + nt * 16 * 64 + head * 32, 64, lane);
        v8f acc = {};
        acc = wmma_f16(a, bf, acc);
        const int j  = nt * 16 + (lane & 15);
        const int mb = (lane & 16) ? 8 : 0;
#pragma unroll
        for (int r = 0; r < 8; ++r) {
            const int i  = mt * 16 + mb + r;
            const int dy = (i >> 3) - (j >> 3) + 7;
            const int dx = (i & 7)  - (j & 7)  + 7;
            sAttn[head * 4096 + i * 64 + j] = acc[r] + rpb[(dy * 15 + dx) * 2 + head];
        }
    }
    __syncthreads();

    // ---- softmax: 128 threads = 128 rows (2 heads x 64 rows)
    {
        const int head = tid >> 6, row = tid & 63;
        float* rp = sAttn + head * 4096 + row * 64;
        float mx = -1e30f;
        for (int j = 0; j < 64; ++j) mx = fmaxf(mx, rp[j]);
        float s = 0.f;
        for (int j = 0; j < 64; ++j) { const float e = __expf(rp[j] - mx); rp[j] = e; s += e; }
        const float inv = 1.f / s;
        _Float16* op = sA16 + head * 4096 + row * 64;
        for (int j = 0; j < 64; ++j) op[j] = (_Float16)(rp[j] * inv);
    }
    __syncthreads();

    // ---- P4: async-stage proj weights, overlapped with AV gemm: attn(64x64) @ v(64x32)
#ifdef HAVE_ASYNC_LDS
    {
        const char* gp = (const char*)projw16;
        for (int i = tid * 16; i < 8192; i += 128 * 16)
            async_b128(gp + i, (char*)sPW + i);
    }
#else
    for (int i = tid; i < 4096; i += 128) sPW[i] = projw16[i];
#endif
    for (int t = wave; t < 16; t += 4) {
        const int head = t & 1, rem = t >> 1;
        const int mt = rem >> 1, nt = rem & 1;
        v8f acc = {};
#pragma unroll
        for (int kc = 0; kc < 64; kc += 32) {
            v16h a  = load_a(sA16 + head * 4096 + mt * 16 * 64 + kc, 64, lane);
            v16h bf = load_b(sV + kc * 64 + head * 32 + nt * 16, 64, lane);
            acc = wmma_f16(a, bf, acc);
        }
        const int col = head * 32 + nt * 16 + (lane & 15);
        const int mb  = (lane & 16) ? 8 : 0;
#pragma unroll
        for (int r = 0; r < 8; ++r)
            sO[(mt * 16 + mb + r) * 64 + col] = (_Float16)acc[r];
    }
    async_wait0();
    __syncthreads();

    // ---- proj gemm + bias + residual into x
    const int Bi = win / 144, wrem = win % 144;
    const int wy = wrem / 12, wx = wrem % 12;
    for (int t = wave; t < 16; t += 4) {
        const int mt = t & 3, nt = t >> 2;
        v8f acc = {};
#pragma unroll
        for (int kc = 0; kc < 64; kc += 32) {
            v16h a  = load_a (sO + mt * 16 * 64 + kc, 64, lane);
            v16h bf = load_bt(sPW + (nt * 16) * 64 + kc, 64, lane);
            acc = wmma_f16(a, bf, acc);
        }
        const int   oc = nt * 16 + (lane & 15);
        const float pb = proj_b[oc];
        const int   mb = (lane & 16) ? 8 : 0;
#pragma unroll
        for (int r = 0; r < 8; ++r) {
            const int n  = mt * 16 + mb + r;
            const int yy = wy * 8 + (n >> 3);
            const int xx = wx * 8 + (n & 7);
            const size_t tok = (size_t)Bi * 9216 + yy * 96 + xx;
            x[tok * 64 + oc] += acc[r] + pb;
        }
    }
}

// ---------------------------------------------------------------------------
// Kernel 3: LN2 + MLP(64->256 GELU ->64) + residual + 1x1 conv(64->32) + leaky.
// One block per 64-token tile (7200 blocks), 128 threads. 48KB LDS.
// fc/conv B-fragments streamed from f16 weights (L2-resident; prefetched).
__global__ __launch_bounds__(128)
void k_mlp(const float* __restrict__ x, const _Float16* __restrict__ w16,
           const float* __restrict__ g2, const float* __restrict__ b2,
           const float* __restrict__ fc1_b, const float* __restrict__ fc2_b,
           const float* __restrict__ conv_b, float* __restrict__ out) {
    __shared__ __align__(16) char smem[49152];
    _Float16* sXn = (_Float16*)smem;             // 64x64 f16 LN2 output
    _Float16* sH  = (_Float16*)(smem + 8192);    // 64x256 f16 hidden
    _Float16* sY  = (_Float16*)(smem + 40960);   // 64x64 f16 pre-conv

    const _Float16* w1 = w16 + 16384;  // fc1_w (256x64)
    const _Float16* w2 = w16 + 32768;  // fc2_w (64x256)
    const _Float16* cw = w16 + 49152;  // conv_w (32x64)

    const int tid  = threadIdx.x, wave = tid >> 5, lane = tid & 31;
    const int tile = blockIdx.x;
    const size_t tbase = (size_t)tile * 64;

    // prefetch weight cachelines (L2-resident across the grid)
    __builtin_prefetch(w1 + tid * 64, 0, 0);
    __builtin_prefetch(w2 + tid * 64, 0, 0);

    // ---- LN2 (64 threads, one token each)
    if (tid < 64) {
        const float* rp = x + (tbase + tid) * 64;
        float s = 0.f;
        for (int k = 0; k < 64; ++k) s += rp[k];
        const float m = s * (1.f / 64.f);
        float v = 0.f;
        for (int k = 0; k < 64; ++k) { const float d = rp[k] - m; v += d * d; }
        const float rs = rsqrtf(v * (1.f / 64.f) + 1e-5f);
        for (int k = 0; k < 64; ++k)
            sXn[tid * 64 + k] = (_Float16)((rp[k] - m) * rs * g2[k] + b2[k]);
    }
    __syncthreads();

    // ---- fc1 + exact GELU: Xn(64x64) @ fc1_w^T(64x256)
    for (int t = wave; t < 64; t += 4) {
        const int mt = t & 3, nt = t >> 2;
        v8f acc = {};
#pragma unroll
        for (int kc = 0; kc < 64; kc += 32) {
            v16h a  = load_a (sXn + mt * 16 * 64 + kc, 64, lane);
            v16h bf = load_bt(w1 + (nt * 16) * 64 + kc, 64, lane);
            acc = wmma_f16(a, bf, acc);
        }
        const int   o    = nt * 16 + (lane & 15);
        const float bias = fc1_b[o];
        const int   mb   = (lane & 16) ? 8 : 0;
#pragma unroll
        for (int r = 0; r < 8; ++r) {
            float v = acc[r] + bias;
            v = 0.5f * v * (1.f + erff(v * 0.70710678118f));
            sH[(mt * 16 + mb + r) * 256 + o] = (_Float16)v;
        }
    }
    __syncthreads();

    // ---- fc2 + residual: H(64x256) @ fc2_w^T(256x64)
    for (int t = wave; t < 16; t += 4) {
        const int mt = t & 3, nt = t >> 2;
        v8f acc = {};
#pragma unroll
        for (int kc = 0; kc < 256; kc += 32) {
            v16h a  = load_a (sH + mt * 16 * 256 + kc, 256, lane);
            v16h bf = load_bt(w2 + (nt * 16) * 256 + kc, 256, lane);
            acc = wmma_f16(a, bf, acc);
        }
        const int   o    = nt * 16 + (lane & 15);
        const float bias = fc2_b[o];
        const int   mb   = (lane & 16) ? 8 : 0;
#pragma unroll
        for (int r = 0; r < 8; ++r) {
            const int row = mt * 16 + mb + r;
            const float v = acc[r] + bias + x[(tbase + row) * 64 + o];
            sY[row * 64 + o] = (_Float16)v;
        }
    }
    __syncthreads();

    // ---- conv 1x1 + leaky: Y(64x64) @ conv_w^T(64x32) -> out (b,an2,c,h,w)
    const int Bi   = tile / 144;
    const int prem = (tile % 144) * 64;
    for (int t = wave; t < 8; t += 4) {
        const int mt = t & 3, nt = t >> 2;
        v8f acc = {};
#pragma unroll
        for (int kc = 0; kc < 64; kc += 32) {
            v16h a  = load_a (sY + mt * 16 * 64 + kc, 64, lane);
            v16h bf = load_bt(cw + (nt * 16) * 64 + kc, 64, lane);
            acc = wmma_f16(a, bf, acc);
        }
        const int   oc   = nt * 16 + (lane & 15);
        const float bias = conv_b[oc];
        const int   mb   = (lane & 16) ? 8 : 0;
#pragma unroll
        for (int r = 0; r < 8; ++r) {
            const int row = mt * 16 + mb + r;
            float v = acc[r] + bias;
            v = (v >= 0.f) ? v : 0.1f * v;
            out[((size_t)Bi * 32 + oc) * 9216 + prem + row] = v;
        }
    }
}

// ---------------------------------------------------------------------------
extern "C" void kernel_launch(void* const* d_in, const int* in_sizes, int n_in,
                              void* d_out, int out_size, void* d_ws, size_t ws_size,
                              hipStream_t stream) {
    const float* gfea   = (const float*)d_in[0];
    const float* cfea   = (const float*)d_in[1];
    const float* disp   = (const float*)d_in[2];
    const float* qkv_w  = (const float*)d_in[3];
    const float* qkv_b  = (const float*)d_in[4];
    const float* proj_w = (const float*)d_in[5];
    const float* proj_b = (const float*)d_in[6];
    const float* rpb    = (const float*)d_in[7];
    const float* g1     = (const float*)d_in[8];
    const float* b1     = (const float*)d_in[9];
    const float* g2     = (const float*)d_in[10];
    const float* b2     = (const float*)d_in[11];
    const float* fc1_w  = (const float*)d_in[12];
    const float* fc1_b  = (const float*)d_in[13];
    const float* fc2_w  = (const float*)d_in[14];
    const float* fc2_b  = (const float*)d_in[15];
    const float* conv_w = (const float*)d_in[16];
    const float* conv_b = (const float*)d_in[17];
    float* out = (float*)d_out;

    char* ws = (char*)d_ws;
    float*    x   = (float*)ws;                                   // 50*9216*64 f32 = 117,964,800 B
    _Float16* xw  = (_Float16*)(ws + 117964800);                  //  58,982,400 B
    _Float16* w16 = (_Float16*)(ws + 117964800 + 58982400);       //     102,400 B

    k_cvt     <<<200,    256, 0, stream>>>(qkv_w, proj_w, fc1_w, fc2_w, conv_w, w16);
    k_warp_ln1<<<460800,  64, 0, stream>>>(gfea, cfea, disp, g1, b1, x, xw);
    k_attn    <<<7200,   128, 0, stream>>>(xw, w16, qkv_b, proj_b, rpb, x);
    k_mlp     <<<7200,   128, 0, stream>>>(x, w16, g2, b2, fc1_b, fc2_b, conv_b, out);
}
